// PHMLinear_54726473286286
// MI455X (gfx1250) — matrix operand order repaired and checked by reading, not compiled, not run on hardware
//
#include <hip/hip_runtime.h>
#include <hip/hip_bf16.h>

// ---------------------------------------------------------------------------
// PHM ensemble linear, fused to: out[8192,1024] = x[8192,1024] @ W[1024,1024]
//   W[(a,i),(o,c)] = sum_{e,r} Wrl[e,i,r]*Wrr[e,r,o]*Wleft[e,a,c]
// Kernel 1: build W in fp32, store bf16 TRANSPOSED (Wt[n][k]) in d_ws (2 MB).
// Kernel 2: bf16 WMMA GEMM (v_wmma_f32_16x16x32_bf16), fp32 accumulate,
//           in-register fp32->bf16 conversion of x, NT stores for out.
// ---------------------------------------------------------------------------

typedef __attribute__((ext_vector_type(16))) __bf16          v16bf;
typedef __attribute__((ext_vector_type(8)))  float           v8f;
typedef __attribute__((ext_vector_type(16))) unsigned short  v16us;
typedef __attribute__((ext_vector_type(8)))  unsigned int    v8u;

#define INDIM  1024
#define OUTDIM 1024
#define MROWS  8192   // B*S = 4*2048

// round-to-nearest-even fp32 -> bf16 (bit pattern) -- scalar fallback
__device__ __forceinline__ unsigned short f2bf(float f) {
    unsigned u = __builtin_bit_cast(unsigned, f);
    unsigned rnd = 0x7FFFu + ((u >> 16) & 1u);
    return (unsigned short)((u + rnd) >> 16);
}

// packed fp32x2 -> bf16x2 in one dword; prefer hardware v_cvt_pk_bf16_f32
__device__ __forceinline__ unsigned pack_bf16(float lo, float hi) {
#if __has_builtin(__builtin_amdgcn_cvt_pk_bf16_f32)
    return __builtin_bit_cast(unsigned,
                              __builtin_amdgcn_cvt_pk_bf16_f32(lo, hi));
#else
    return (unsigned)f2bf(lo) | ((unsigned)f2bf(hi) << 16);
#endif
}

// ---------------------------------------------------------------------------
// Kernel 1: combined weight, bf16, transposed: Wt[n*1024 + k], n=(o,c), k=(a,i)
// 1M threads, ~40 MACs each; all weight tensors (~34 KB) stay in cache.
// ---------------------------------------------------------------------------
__global__ __launch_bounds__(256)
void phm_build_wt(const float* __restrict__ wrl,   // [8,128,4]
                  const float* __restrict__ wrr,   // [8,4,128]
                  const float* __restrict__ wl,    // [8,8,8]
                  unsigned short* __restrict__ wt) // [1024 n][1024 k] bf16
{
    int idx = blockIdx.x * blockDim.x + threadIdx.x;   // 0 .. 1024*1024-1
    int n = idx >> 10;        // output index (o,c):  n = o*8 + c
    int k = idx & 1023;       // input  index (a,i):  k = a*128 + i
    int a = k >> 7, i = k & 127;
    int o = n >> 3, c = n & 7;

    float acc = 0.f;
#pragma unroll
    for (int e = 0; e < 8; ++e) {
        float m2 = 0.f;
#pragma unroll
        for (int r = 0; r < 4; ++r)
            m2 += wrl[(e * 128 + i) * 4 + r] * wrr[(e * 4 + r) * 128 + o];
        acc += m2 * wl[(e * 8 + a) * 8 + c];
    }
    wt[(size_t)n * INDIM + k] = f2bf(acc);
}

// ---------------------------------------------------------------------------
// Kernel 2: GEMM. Block = 8 waves (32x8). Each wave -> one 16(M) x 64(N) strip.
// Grid: x = N strips (1024/64 = 16), y = M tile groups (8192/16/8 = 64).
// K loop: 1024 in steps of 32 (one v_wmma_f32_16x16x32_bf16 per N-subtile).
// ---------------------------------------------------------------------------
__global__ __launch_bounds__(256)
void phm_gemm_bf16(const float* __restrict__ x,            // [8192,1024] fp32
                   const unsigned short* __restrict__ wt,  // [1024,1024] bf16 (transposed)
                   float* __restrict__ out)                // [8192,1024] fp32
{
    const int lane = threadIdx.x;          // 0..31
    const int wave = threadIdx.y;          // 0..7
    const int l15  = lane & 15;
    const int half = lane >> 4;            // 0: lanes 0-15, 1: lanes 16-31

    const int m0 = (blockIdx.y * 8 + wave) * 16;   // M tile origin
    const int n0 = blockIdx.x * 64;                // N strip origin

    // A-fragment addressing (16-bit A 16x32, contiguous K to match B):
    //   lanes 0-15 : row M=lane,    elems 0..15 = K 0..15
    //   lanes 16-31: row M=lane-16, elems 0..15 = K 16..31
    const float* xrow = x + (size_t)(m0 + l15) * INDIM;
    const int koA = half ? 16 : 0;
    // B-fragment addressing (16-bit B 32x16): col N = lane&15,
    //   lanes 0-15 hold K 0..15, lanes 16-31 hold K 16..31 (contiguous in Wt)
    const int koB = half ? 16 : 0;
    const unsigned short* wcol[4];
#pragma unroll
    for (int t = 0; t < 4; ++t)
        wcol[t] = wt + (size_t)(n0 + t * 16 + l15) * INDIM + koB;

    v8f acc[4] = {};   // 4 N-subtiles of 16x16 fp32 accumulators

    for (int kk = 0; kk < INDIM; kk += 32) {
        // speculative prefetch ~2KB ahead; OOB translation fails are dropped
        __builtin_prefetch(xrow + kk + 512, 0, 1);

        // issue A loads + all 4 B-fragment loads before any WMMA
        float4 a0 = *(const float4*)(xrow + kk + koA);
        float4 a1 = *(const float4*)(xrow + kk + koA + 4);
        float4 a2 = *(const float4*)(xrow + kk + koA + 8);
        float4 a3 = *(const float4*)(xrow + kk + koA + 12);
        v16us bu[4];
#pragma unroll
        for (int t = 0; t < 4; ++t)
            bu[t] = *(const v16us*)(wcol[t] + kk);   // 32B contiguous, 32B aligned

        // convert A fragment (16 fp32 -> 8 packed bf16 dwords per lane)
        v8u au;
        au[0] = pack_bf16(a0.x, a0.y);  au[1] = pack_bf16(a0.z, a0.w);
        au[2] = pack_bf16(a1.x, a1.y);  au[3] = pack_bf16(a1.z, a1.w);
        au[4] = pack_bf16(a2.x, a2.y);  au[5] = pack_bf16(a2.z, a2.w);
        au[6] = pack_bf16(a3.x, a3.y);  au[7] = pack_bf16(a3.z, a3.w);
        v16bf afrag = __builtin_bit_cast(v16bf, au);

#pragma unroll
        for (int t = 0; t < 4; ++t) {
            v16bf bfrag = __builtin_bit_cast(v16bf, bu[t]);
            // D = A*B + C : emits v_wmma_f32_16x16x32_bf16
            acc[t] = __builtin_amdgcn_wmma_f32_16x16x32_bf16(
                         /*neg_a=*/false, afrag,
                         /*neg_b=*/false, bfrag,
                         /*c_mod=*/(short)0, acc[t],
                         /*reuse_a=*/false, /*reuse_b=*/false);
        }
    }

    // store: C/D layout — VGPR v: lanes 0-15 -> M=m0+v, lanes 16-31 -> M=m0+8+v
    const int row_base = m0 + (half ? 8 : 0);
#pragma unroll
    for (int t = 0; t < 4; ++t) {
        const int n = n0 + t * 16 + l15;
#pragma unroll
        for (int v = 0; v < 8; ++v) {
            __builtin_nontemporal_store(acc[t][v],
                out + (size_t)(row_base + v) * OUTDIM + n);
        }
    }
}

// ---------------------------------------------------------------------------
extern "C" void kernel_launch(void* const* d_in, const int* in_sizes, int n_in,
                              void* d_out, int out_size, void* d_ws, size_t ws_size,
                              hipStream_t stream) {
    (void)in_sizes; (void)n_in; (void)out_size; (void)ws_size;
    const float* x   = (const float*)d_in[0];   // [4,2048,1024]
    const float* wrl = (const float*)d_in[1];   // [8,128,4]
    const float* wrr = (const float*)d_in[2];   // [8,4,128]
    const float* wl  = (const float*)d_in[3];   // [8,8,8]
    float* out = (float*)d_out;                 // [4,2048,1024]
    unsigned short* wt = (unsigned short*)d_ws; // 1024*1024 bf16 = 2 MB scratch

    // 1) build combined bf16 transposed weight (deterministic, rebuilt each call)
    phm_build_wt<<<dim3(1024 * 1024 / 256), dim3(256), 0, stream>>>(wrl, wrr, wl, wt);

    // 2) WMMA GEMM: grid (N strips, M groups), block (32 lanes, 8 waves)
    phm_gemm_bf16<<<dim3(OUTDIM / 64, MROWS / (16 * 8)), dim3(32, 8), 0, stream>>>(
        x, wt, out);
}
